// EncoderwithProjection_11922829214128
// MI455X (gfx1250) — compile-verified, matrix operand order checked by reading, not run on hardware
//
#include <hip/hip_runtime.h>
#include <hip/hip_bf16.h>

// ---------------------------------------------------------------------------
// Problem constants (from reference)
// ---------------------------------------------------------------------------
constexpr int BS   = 256;
constexpr int EMB  = 768;
constexpr int HW   = 196;            // 14*14
constexpr int NM   = 16;             // n_masks
constexpr int HID  = 4096;
constexpr int OUTD = 256;
constexpr int NP   = 4;
constexpr int ROWS = BS * NM;        // 4096 pooled rows
constexpr float BN_EPS = 1e-5f;

constexpr int KB1 = EMB / 32;        // 24  k-blocks for GEMM1
constexpr int NB1 = HID / 16;        // 256 n-blocks for GEMM1
constexpr int KB2 = HID / 32;        // 128 k-blocks for GEMM2
constexpr int NB2 = OUTD / 16;       // 16  n-blocks for GEMM2

typedef __attribute__((ext_vector_type(16))) _Float16 v16h;
typedef __attribute__((ext_vector_type(8)))  _Float16 v8h;
typedef __attribute__((ext_vector_type(8)))  float    v8f;
typedef __attribute__((ext_vector_type(4)))  int      v4i;

// ---------------------------------------------------------------------------
// Async global->LDS staging (gfx1250 GLOBAL_LOAD_ASYNC_TO_LDS_B128, ASYNCcnt)
// with a portable fallback. ROCm 7.2 signature takes v4i pointers in
// addrspace(1)/(3).
// ---------------------------------------------------------------------------
#if defined(__HIP_DEVICE_COMPILE__) &&                                        \
    __has_builtin(__builtin_amdgcn_global_load_async_to_lds_b128) &&          \
    __has_builtin(__builtin_amdgcn_s_wait_asynccnt)
#define USE_ASYNC_LDS 1
#else
#define USE_ASYNC_LDS 0
#endif

__device__ __forceinline__ void stage16(const _Float16* src, _Float16* dst) {
#if USE_ASYNC_LDS
    __builtin_amdgcn_global_load_async_to_lds_b128(
        (__attribute__((address_space(1))) v4i*)(src),
        (__attribute__((address_space(3))) v4i*)(dst),
        /*offset=*/0, /*cpol=*/0);
#else
    *(v8h*)dst = *(const v8h*)src;
#endif
}

__device__ __forceinline__ void stage_wait() {
#if USE_ASYNC_LDS
    __builtin_amdgcn_s_wait_asynccnt(0);
#endif
}

__device__ __forceinline__ v8f wmma_f16(v16h a, v16h b, v8f c) {
    // D = A(16x32 f16) * B(32x16 f16) + C(16x16 f32)
    return __builtin_amdgcn_wmma_f32_16x16x32_f16(
        /*neg_a=*/false, a, /*neg_b=*/false, b,
        /*c_mod=*/(short)0, c, /*reuse_a=*/false, /*reuse_b=*/false);
}

// A-fragment (16x32, f16) from row-major f16 matrix.
// ISA layout: lane L holds row M=L%16; half j -> K = j + 8*(j>>3) + 8*(L>>4).
// For fixed lane this is two contiguous 8-half (16B) runs.
__device__ __forceinline__ v16h load_a_rowmajor(const _Float16* base, int ldk, int kb) {
    int lane = threadIdx.x & 31;
    int m  = lane & 15;
    int hi = lane >> 4;
    const _Float16* p = base + (size_t)m * ldk + kb * 32 + hi * 8;
    v8h lo = *(const v8h*)(p);
    v8h hh = *(const v8h*)(p + 16);
    v16h a;
#pragma unroll
    for (int i = 0; i < 8; ++i) { a[i] = lo[i]; a[8 + i] = hh[i]; }
    return a;
}

// ---------------------------------------------------------------------------
// Pack W1 (P,EMB,HID) f32 -> B-fragments f16.
// B (32x16, KxN) assumed mirror of A: lane L holds col N=L%16,
// half j -> K = (j&7) + 16*(j>>3) + 8*(L>>4).
// Fragment f = (p*KB1 + kb)*NB1 + nb, 512 halves each (lane-major, contiguous).
// ---------------------------------------------------------------------------
__global__ __launch_bounds__(256) void pack_w1_kernel(const float* __restrict__ W1,
                                                      _Float16* __restrict__ W1t) {
    int idx  = blockIdx.x * 256 + threadIdx.x;       // NP*KB1*NB1*512 total
    int j    = idx & 15;
    int lane = (idx >> 4) & 31;
    int frag = idx >> 9;
    int nb = frag % NB1;
    int kb = (frag / NB1) % KB1;
    int p  = frag / (NB1 * KB1);
    int hi = lane >> 4;
    int nn = nb * 16 + (lane & 15);
    int k  = kb * 32 + (j & 7) + ((j >> 3) << 4) + 8 * hi;
    W1t[idx] = (_Float16)W1[((size_t)p * EMB + k) * HID + nn];
}

__global__ __launch_bounds__(256) void pack_w2_kernel(const float* __restrict__ W2,
                                                      _Float16* __restrict__ W2t) {
    int idx  = blockIdx.x * 256 + threadIdx.x;       // NP*KB2*NB2*512 total
    int j    = idx & 15;
    int lane = (idx >> 4) & 31;
    int frag = idx >> 9;
    int nb = frag % NB2;
    int kb = (frag / NB2) % KB2;
    int p  = frag / (NB2 * KB2);
    int hi = lane >> 4;
    int oo = nb * 16 + (lane & 15);
    int k  = kb * 32 + (j & 7) + ((j >> 3) << 4) + 8 * hi;
    W2t[idx] = (_Float16)W2[((size_t)p * HID + k) * OUTD + oo];
}

// ---------------------------------------------------------------------------
// Mask-normalized pooling: pooled[b*NM+n][e] f16.
// One block per batch element; masks staged in LDS.
// ---------------------------------------------------------------------------
__global__ __launch_bounds__(256) void pool_kernel(const float* __restrict__ x,
                                                   const float* __restrict__ masks,
                                                   _Float16* __restrict__ pooled) {
    __shared__ float mLds[NM * HW];
    __shared__ float invArea[NM];
    int b   = blockIdx.x;
    int tid = threadIdx.x;
    for (int i = tid; i < NM * HW; i += 256)
        mLds[i] = masks[(size_t)b * NM * HW + i];
    __syncthreads();
    if (tid < NM) {
        float a = 0.f;
        for (int hw = 0; hw < HW; ++hw) a += mLds[tid * HW + hw];
        invArea[tid] = 1.0f / fmaxf(a, 1.0f);
    }
    __syncthreads();
    for (int e = tid; e < EMB; e += 256) {
        const float* xp = x + ((size_t)b * EMB + e) * HW;
        float acc[NM];
#pragma unroll
        for (int n = 0; n < NM; ++n) acc[n] = 0.f;
        for (int hw = 0; hw < HW; ++hw) {
            float xv = xp[hw];
#pragma unroll
            for (int n = 0; n < NM; ++n) acc[n] += mLds[n * HW + hw] * xv;
        }
#pragma unroll
        for (int n = 0; n < NM; ++n)
            pooled[((size_t)(b * NM + n)) * EMB + e] = (_Float16)(acc[n] * invArea[n]);
    }
}

// ---------------------------------------------------------------------------
// GEMM1 + BN statistics: h = pooled @ W1[p] + b1[p]; accumulate per-column
// sum / sumsq over all 4096 rows via atomics. h itself is discarded.
// grid = (NB1, ROWS/(16*8), NP), 8 waves per block, one 16x16 tile per wave.
// The 24KB B-fragment column for (p, nb) is staged once per block into LDS
// via async global->LDS copies and shared by all 8 waves.
// ---------------------------------------------------------------------------
__global__ __launch_bounds__(256) void gemm1_stats_kernel(const _Float16* __restrict__ pooled,
                                                          const _Float16* __restrict__ W1t,
                                                          const float* __restrict__ b1,
                                                          float* __restrict__ sum,
                                                          float* __restrict__ sumsq) {
    __shared__ __align__(32) _Float16 bLds[KB1 * 512];   // 24 KB
    int nb   = blockIdx.x;
    int p    = blockIdx.z;
    int tid  = threadIdx.x;
    int wave = tid >> 5;
    int lane = tid & 31;
    int mb   = blockIdx.y * 8 + wave;

    // Stage B fragments (all 24 kb for this nb): 1536 x 16B chunks, 6 per thread.
    for (int i = tid; i < KB1 * 64; i += 256) {
        int kb  = i >> 6;
        int off = (i & 63) * 8;
        const _Float16* src = W1t + (((size_t)(p * KB1 + kb) * NB1 + nb) << 9) + off;
        stage16(src, bLds + kb * 512 + off);
    }
    stage_wait();
    __syncthreads();

    const _Float16* aBase = pooled + (size_t)(mb * 16) * EMB;
    v8f acc = {};
#pragma unroll 4
    for (int kb = 0; kb < KB1; ++kb) {
        v16h a = load_a_rowmajor(aBase, EMB, kb);
        v16h b = *(const v16h*)(bLds + kb * 512 + lane * 16);
        acc = wmma_f16(a, b, acc);
    }
    // C/D layout: lane holds col N = lane&15, rows M = r + 8*(lane>>4)
    int col    = nb * 16 + (lane & 15);
    float bias = b1[p * HID + col];
    float s = 0.f, q = 0.f;
#pragma unroll
    for (int r = 0; r < 8; ++r) {
        float v = acc[r] + bias;
        s += v;
        q += v * v;
    }
    s += __shfl_down(s, 16, 32);
    q += __shfl_down(q, 16, 32);
    if (lane < 16) {
        atomicAdd(&sum[p * HID + col], s);
        atomicAdd(&sumsq[p * HID + col], q);
    }
}

__global__ __launch_bounds__(256) void bn_finalize_kernel(const float* __restrict__ sum,
                                                          const float* __restrict__ sumsq,
                                                          const float* __restrict__ gamma,
                                                          const float* __restrict__ beta,
                                                          float* __restrict__ bnScale,
                                                          float* __restrict__ bnShift) {
    int i = blockIdx.x * 256 + threadIdx.x;
    if (i >= NP * HID) return;
    float mu  = sum[i] * (1.0f / ROWS);
    float var = sumsq[i] * (1.0f / ROWS) - mu * mu;
    float sc  = gamma[i] * rsqrtf(fmaxf(var, 0.f) + BN_EPS);
    bnScale[i] = sc;
    bnShift[i] = beta[i] - mu * sc;
}

// ---------------------------------------------------------------------------
// Fused: recompute h rows for the SELECTED projector only, BN+ReLU, GEMM2.
// grid = (BS/16 b-tiles, NM), 256 threads (8 waves).
// Per 128-wide hid chunk: wave w computes h[16, w*16..w*16+16) (24 WMMAs),
// BN+ReLU, stores to LDS in A-fragment layout, then all waves run GEMM2
// k-steps (4 per chunk) into their two persistent 16x16 output accumulators.
// ---------------------------------------------------------------------------
__global__ __launch_bounds__(256) void fused_bn_gemm2_kernel(const _Float16* __restrict__ pooled,
                                                             const _Float16* __restrict__ W1t,
                                                             const _Float16* __restrict__ W2t,
                                                             const float* __restrict__ b1,
                                                             const float* __restrict__ bnScale,
                                                             const float* __restrict__ bnShift,
                                                             const float* __restrict__ b2,
                                                             const int* __restrict__ projIdx,
                                                             float* __restrict__ out) {
    __shared__ __align__(32) _Float16 aLds[KB1 * 512];  // 24 KB: pooled tile as A-frags
    __shared__ __align__(32) _Float16 hLds[4 * 512];    // 4 KB: BN'd h chunk as A-frags

    int btile = blockIdx.x;          // 0..15
    int n     = blockIdx.y;          // 0..15
    int p     = projIdx[n];
    int tid   = threadIdx.x;
    int wave  = tid >> 5;
    int lane  = tid & 31;

    // Stage the 16x768 pooled tile (rows b = btile*16+m, slot n) into LDS in
    // A-fragment layout. Each 16B global run maps to a 16B LDS run -> async.
    for (int i = tid; i < KB1 * 64; i += 256) {
        int kb = i >> 6;
        int r  = i & 63;
        int ln = r >> 1;             // fragment lane 0..31
        int s  = r & 1;              // low/high 8-half run
        int m  = ln & 15;
        int hi = ln >> 4;
        int row = (btile * 16 + m) * NM + n;
        const _Float16* src = pooled + (size_t)row * EMB + kb * 32 + 8 * hi + 16 * s;
        stage16(src, aLds + kb * 512 + ln * 16 + 8 * s);
    }
    stage_wait();
    __syncthreads();

    v8f o0 = {}, o1 = {};            // two output tiles per wave: nt = 2w, 2w+1
    int nt0 = wave * 2, nt1 = wave * 2 + 1;

    for (int chunk = 0; chunk < 32; ++chunk) {
        // ---- recompute h tile: cols [chunk*128 + wave*16, +16) over K=768
        int hidBase = chunk * 128 + wave * 16;
        int nbW1    = hidBase >> 4;
        v8f hacc = {};
#pragma unroll 4
        for (int kb = 0; kb < KB1; ++kb) {
            v16h a = *(const v16h*)(aLds + kb * 512 + lane * 16);
            v16h b = *(const v16h*)(W1t + (((size_t)(p * KB1 + kb) * NB1 + nbW1) << 9) + lane * 16);
            hacc = wmma_f16(a, b, hacc);
        }
        int col  = hidBase + (lane & 15);
        float bi = b1[p * HID + col];
        float sc = bnScale[p * HID + col];
        float sh = bnShift[p * HID + col];
        _Float16 hv[8];
#pragma unroll
        for (int r = 0; r < 8; ++r) {
            float v = (hacc[r] + bi) * sc + sh;
            hv[r] = (_Float16)(v > 0.f ? v : 0.f);
        }

        __syncthreads();             // prior chunk's GEMM2 reads are done
        // scatter into A-fragment layout: element (M, k2), k2 in [0,128)
        {
            int k2     = wave * 16 + (lane & 15);
            int kb2    = k2 >> 5;
            int klocal = k2 & 31;
            int hi2    = (klocal >> 3) & 1;
            int j2     = (klocal & 7) + 8 * (klocal >> 4);
#pragma unroll
            for (int r = 0; r < 8; ++r) {
                int m = r + 8 * (lane >> 4);
                hLds[kb2 * 512 + (m + 16 * hi2) * 16 + j2] = hv[r];
            }
        }
        __syncthreads();

        // ---- GEMM2 partial: 4 k-steps of 32 over this 128-wide chunk
#pragma unroll
        for (int kk = 0; kk < 4; ++kk) {
            v16h a  = *(const v16h*)(hLds + kk * 512 + lane * 16);
            size_t fb = ((size_t)(p * KB2 + chunk * 4 + kk) * NB2) << 9;
            v16h bA = *(const v16h*)(W2t + fb + ((size_t)nt0 << 9) + lane * 16);
            v16h bB = *(const v16h*)(W2t + fb + ((size_t)nt1 << 9) + lane * 16);
            o0 = wmma_f16(a, bA, o0);
            o1 = wmma_f16(a, bB, o1);
        }
    }

    // Epilogue: + b2, write xs[b][n][o]
#pragma unroll
    for (int t = 0; t < 2; ++t) {
        v8f acc = t ? o1 : o0;
        int nt  = t ? nt1 : nt0;
        int oc  = nt * 16 + (lane & 15);
        float bias2 = b2[p * OUTD + oc];
#pragma unroll
        for (int r = 0; r < 8; ++r) {
            int m = r + 8 * (lane >> 4);
            int b = btile * 16 + m;
            out[((size_t)(b * NM + n)) * OUTD + oc] = acc[r] + bias2;
        }
    }
}

// ---------------------------------------------------------------------------
// Host launcher
// ---------------------------------------------------------------------------
extern "C" void kernel_launch(void* const* d_in, const int* in_sizes, int n_in,
                              void* d_out, int out_size, void* d_ws, size_t ws_size,
                              hipStream_t stream) {
    const float* x      = (const float*)d_in[0];   // [256,768,14,14]
    const float* masks  = (const float*)d_in[1];   // [256,16,196]
    const int*   pidx   = (const int*)d_in[2];     // [16]
    const int*   midv   = (const int*)d_in[3];     // [256,16]
    const float* W1     = (const float*)d_in[4];   // [4,768,4096]
    const float* b1     = (const float*)d_in[5];   // [4,4096]
    const float* gamma  = (const float*)d_in[6];   // [4,4096]
    const float* beta   = (const float*)d_in[7];   // [4,4096]
    const float* W2     = (const float*)d_in[8];   // [4,4096,256]
    const float* b2     = (const float*)d_in[9];   // [4,256]
    float* out = (float*)d_out;

    // Workspace layout (bytes)
    char* ws = (char*)d_ws;
    size_t off = 0;
    _Float16* pooled = (_Float16*)(ws + off); off += (size_t)ROWS * EMB * 2;            // 6 MB
    _Float16* W1t    = (_Float16*)(ws + off); off += (size_t)NP * KB1 * NB1 * 512 * 2;  // 24 MB
    _Float16* W2t    = (_Float16*)(ws + off); off += (size_t)NP * KB2 * NB2 * 512 * 2;  // 8 MB
    float* statSum   = (float*)(ws + off);    off += (size_t)NP * HID * 4;
    float* statSq    = (float*)(ws + off);    off += (size_t)NP * HID * 4;
    float* bnScale   = (float*)(ws + off);    off += (size_t)NP * HID * 4;
    float* bnShift   = (float*)(ws + off);    off += (size_t)NP * HID * 4;
    (void)ws_size; (void)in_sizes; (void)n_in; (void)out_size;

    // Zero the stat accumulators (graph-capture-safe)
    (void)hipMemsetAsync(statSum, 0, (size_t)NP * HID * 4 * 2, stream);

    // Pack weights to f16 fragment buffers
    pack_w1_kernel<<<(NP * KB1 * NB1 * 512) / 256, 256, 0, stream>>>(W1, W1t);
    pack_w2_kernel<<<(NP * KB2 * NB2 * 512) / 256, 256, 0, stream>>>(W2, W2t);

    // Mask-normalized pooling
    pool_kernel<<<BS, 256, 0, stream>>>(x, masks, pooled);

    // GEMM1 (all projectors) + BN statistics
    gemm1_stats_kernel<<<dim3(NB1, ROWS / (16 * 8), NP), 256, 0, stream>>>(
        pooled, W1t, b1, statSum, statSq);

    // BN finalize
    bn_finalize_kernel<<<(NP * HID + 255) / 256, 256, 0, stream>>>(
        statSum, statSq, gamma, beta, bnScale, bnShift);

    // Recompute selected-projector h, BN+ReLU, GEMM2, write xs
    fused_bn_gemm2_kernel<<<dim3(BS / 16, NM), 256, 0, stream>>>(
        pooled, W1t, W2t, b1, bnScale, bnShift, b2, pidx, out);

    // mask_ids passthrough (second tuple output, int32 bits)
    (void)hipMemcpyAsync(out + (size_t)BS * NM * OUTD, midv, (size_t)BS * NM * 4,
                         hipMemcpyDeviceToDevice, stream);
}
